// FBPINN_fast_48103633715351
// MI455X (gfx1250) — matrix operand order repaired and checked by reading, not compiled, not run on hardware
//
#include <hip/hip_runtime.h>
#include <hip/hip_bf16.h>

// ---------------------------------------------------------------------------
// FBPINN fused evaluation for MI455X (gfx1250, wave32, WMMA).
//
// S=64 subdomains, N=131072 points, D=2, H=32.
// Each wave owns TWO 16-point tiles; activations are WMMA B-operands
// (32 hid x 16 pts), weights the A-operands (shared by both tiles).
// Hidden layers: v_wmma_f32_16x16x32_f16, 8 per subdomain iteration.
//
// - A-operand rows PERMUTED (A1 = W rows [0..7,16..23], A2 = [8..15,24..31])
//   so D layout == next layer's B layout: repack is pure in-lane
//   v_tanh_f32 + v_cvt_pk_f16_f32 (no shuffles/selects).
// - Double-buffered LDS staging, s-loop unrolled x2 so buffer indices are
//   compile-time constants (immediate DS offsets, no address recompute).
// - 512-thread blocks: staged weights serve 512 points, halving L2 traffic.
// - Window sigmoids split across half-waves (only hh==0 lanes consume acc).
// ---------------------------------------------------------------------------

typedef __attribute__((ext_vector_type(16))) _Float16 v16h;
typedef __attribute__((ext_vector_type(8)))  float    v8f;
typedef __attribute__((ext_vector_type(4)))  int      v4i;

#define N_SUB   64
#define HID     32
#define TPW     2            // tiles (of 16 points) per wave
#define BLOCKT  512          // threads per block (16 waves)

__device__ __forceinline__ float frcp(float x) {
    return __builtin_amdgcn_rcpf(x);           // v_rcp_f32
}

__device__ __forceinline__ float ftanh(float x) {
#if __has_builtin(__builtin_amdgcn_tanhf)
    return __builtin_amdgcn_tanhf(x);          // v_tanh_f32 (gfx1250 native)
#else
    float e = __expf(2.0f * x);
    return 1.0f - 2.0f * frcp(e + 1.0f);
#endif
}

__device__ __forceinline__ float fsig(float z) {
    return frcp(1.0f + __expf(-z));            // v_exp_f32 + v_rcp_f32
}

__device__ __forceinline__ unsigned pk2(float a, float b) {
    union { _Float16 h[2]; unsigned u; } q;
    q.h[0] = (_Float16)a;                      // v_cvt_pk_f16_f32
    q.h[1] = (_Float16)b;
    return q.u;
}

union V16 { v16h v; v4i q[2]; };
union V8  { v8f  v; v4i q[2]; };

__global__ void __launch_bounds__(BLOCKT)
fbpinn_fused_wmma(const float* __restrict__ x,
                  const float* __restrict__ lo_core,
                  const float* __restrict__ hi_core,
                  const float* __restrict__ lo_ext,
                  const float* __restrict__ hi_ext,
                  const float* __restrict__ W_in,
                  const float* __restrict__ b_in,
                  const float* __restrict__ W_h1,
                  const float* __restrict__ b_h1,
                  const float* __restrict__ W_h2,
                  const float* __restrict__ b_h2,
                  const float* __restrict__ W_out,
                  const float* __restrict__ b_out,
                  float* __restrict__ out,
                  int n_pts) {
    // ---- double-buffered LDS staging (~10 KB) ----
    __shared__ alignas(16) _Float16 sWh1[2][HID * HID];
    __shared__ alignas(16) _Float16 sWh2[2][HID * HID];
    __shared__ alignas(16) float    sWin[2][HID * 2];
    __shared__ alignas(16) float    sBin[2][HID];
    __shared__ alignas(16) float    sB1[2][HID];
    __shared__ alignas(16) float    sB2[2][HID];
    __shared__ alignas(16) float    sWout[2][HID];
    __shared__ float sGeom[2][12];

    const int tid  = threadIdx.x;
    const int lane = tid & 31;
    const int col  = lane & 15;     // point column / A-matrix row id
    const int hh   = lane >> 4;     // half-wave id
    const int wave = tid >> 5;
    const int gw   = blockIdx.x * (BLOCKT >> 5) + wave;

    // Two 16-point tiles per wave
    int   pp[TPW];
    bool  pv[TPW];
    float xt[TPW], xxv[TPW];
#pragma unroll
    for (int t = 0; t < TPW; ++t) {
        const int p = gw * (16 * TPW) + 16 * t + col;
        pv[t] = (p < n_pts);
        pp[t] = p;
        const int pc = pv[t] ? p : (n_pts - 1);
        xt[t]  = x[2 * pc + 0];   // t-coordinate
        xxv[t] = x[2 * pc + 1];   // x-coordinate
    }

    // Permuted weight-row indices: A1 = W rows [0..7,16..23], A2 = +8
    const int wrow1 = col + ((col >= 8) ? 8 : 0);
    const int wrow2 = wrow1 + 8;

    // coordinates whose window THIS lane computes (tile hh)
    const float wxt = hh ? xt[1] : xt[0];
    const float wxx = hh ? xxv[1] : xxv[0];

    float acc[TPW]  = {0.0f, 0.0f};
    float wacc[TPW] = {0.0f, 0.0f};

    // ---- staging: subdomain s -> buffer b (b is a literal at call sites) ----
    auto stage = [&](int s, int b) {
        // 1024 f32 per matrix, 512 threads -> one float2 each
        const float2 v1 = ((const float2*)(W_h1 + s * HID * HID))[tid];
        const float2 v2 = ((const float2*)(W_h2 + s * HID * HID))[tid];
        ((unsigned*)&sWh1[b][0])[tid] = pk2(v1.x, v1.y);   // ds_store_b32
        ((unsigned*)&sWh2[b][0])[tid] = pk2(v2.x, v2.y);
        if (tid < HID * 2) sWin[b][tid] = W_in[s * HID * 2 + tid];
        if (tid < HID) {
            sBin[b][tid]  = b_in[s * HID + tid];
            sB1[b][tid]   = b_h1[s * HID + tid];
            sB2[b][tid]   = b_h2[s * HID + tid];
            sWout[b][tid] = W_out[s * HID + tid];
        }
        if (tid < 2) {
            const int d = tid;
            float lc = lo_core[s * 2 + d], hc = hi_core[s * 2 + d];
            float le = lo_ext[s * 2 + d],  he = hi_ext[s * 2 + d];
            sGeom[b][0 + d] = lc;
            sGeom[b][2 + d] = hc;
            sGeom[b][4 + d] = (le + he) * 0.5f;
            sGeom[b][6 + d] = frcp((he - le) * 0.5f);
            float overlap = fmaxf(he - hc, lc - le);
            sGeom[b][8 + d] = 4.0f * frcp(2.0f * overlap * (he - le) + 1e-8f);
        }
        if (tid == 0) sGeom[b][10] = b_out[s];
    };

    // ---- full per-subdomain compute from buffer b (b literal -> const DS offsets)
    auto compute = [&](int b) {
        const float lc0 = sGeom[b][0], lc1 = sGeom[b][1];
        const float hc0 = sGeom[b][2], hc1 = sGeom[b][3];
        const float c0  = sGeom[b][4], c1  = sGeom[b][5];
        const float ih0 = sGeom[b][6], ih1 = sGeom[b][7];
        const float ss0 = sGeom[b][8], ss1 = sGeom[b][9];
        const float bO  = sGeom[b][10];

        // window: each half-wave computes its own tile's window
        float wm = fsig(ss0 * (wxt - lc0)) * fsig(ss0 * (hc0 - wxt))
                 * fsig(ss1 * (wxx - lc1)) * fsig(ss1 * (hc1 - wxx));
        float wo = __shfl_xor(wm, 16, 32);
        float w[TPW];
        w[0] = wm;   // correct on lower lanes (the only consumers of acc)
        w[1] = wo;

        // layer 1 (K=2, VALU) -> B operands; weights loaded once per j
        float xn0[TPW], xn1[TPW];
#pragma unroll
        for (int t = 0; t < TPW; ++t) {
            xn0[t] = (xt[t]  - c0) * ih0;
            xn1[t] = (xxv[t] - c1) * ih1;
        }
        v16h bop[TPW];
#pragma unroll
        for (int j = 0; j < 16; ++j) {
            const int u = j + (hh << 4);
            const float w0 = sWin[b][2 * u];
            const float w1 = sWin[b][2 * u + 1];
            const float bi = sBin[b][u];
#pragma unroll
            for (int t = 0; t < TPW; ++t) {
                float val = fmaf(xn1[t], w1, fmaf(xn0[t], w0, bi));
                bop[t][j] = (_Float16)ftanh(val);
            }
        }

        // A operands (contiguous k = 16*hh + s), shared by tiles
        const int coff = hh ? 32 : 0;
        auto loadA = [&](const _Float16* base, int row) -> v16h {
            const char* pr = (const char*)base + row * 64 + coff;
            V16 a;
            a.q[0] = *(const v4i*)(pr);
            a.q[1] = *(const v4i*)(pr + 16);
            return a.v;
        };
        const v16h a1l = loadA(&sWh1[b][0], wrow1);
        const v16h a1h = loadA(&sWh1[b][0], wrow2);
        const v16h a2l = loadA(&sWh2[b][0], wrow1);
        const v16h a2h = loadA(&sWh2[b][0], wrow2);

        // biases as C operands, permuted to match A-row ordering
        auto loadC = [&](const float* bb, int off) -> v8f {
            const char* pb = (const char*)bb + off * 4;
            V8 c;
            c.q[0] = *(const v4i*)(pb);
            c.q[1] = *(const v4i*)(pb + 16);
            return c.v;
        };
        const v8f c1l = loadC(&sB1[b][0], 16 * hh);
        const v8f c1h = loadC(&sB1[b][0], 8 + 16 * hh);
        const v8f c2l = loadC(&sB2[b][0], 16 * hh);
        const v8f c2h = loadC(&sB2[b][0], 8 + 16 * hh);

        // hidden layer 1: 4 WMMAs, tiles interleaved
        v8f d1[TPW], d2[TPW];
#pragma unroll
        for (int t = 0; t < TPW; ++t) {
            d1[t] = __builtin_amdgcn_wmma_f32_16x16x32_f16(false, a1l, false, bop[t],
                                                           (short)0, c1l, false, false);
            d2[t] = __builtin_amdgcn_wmma_f32_16x16x32_f16(false, a1h, false, bop[t],
                                                           (short)0, c1h, false, false);
        }

        // tanh + D->B repack: pure in-lane
#pragma unroll
        for (int t = 0; t < TPW; ++t) {
#pragma unroll
            for (int r = 0; r < 8; ++r) {
                bop[t][r]     = (_Float16)ftanh(d1[t][r]);
                bop[t][8 + r] = (_Float16)ftanh(d2[t][r]);
            }
        }

        // hidden layer 2: 4 WMMAs
#pragma unroll
        for (int t = 0; t < TPW; ++t) {
            d1[t] = __builtin_amdgcn_wmma_f32_16x16x32_f16(false, a2l, false, bop[t],
                                                           (short)0, c2l, false, false);
            d2[t] = __builtin_amdgcn_wmma_f32_16x16x32_f16(false, a2h, false, bop[t],
                                                           (short)0, c2h, false, false);
        }

        // output layer: dot(tanh(h3), W_out) on the permuted D layout
#pragma unroll
        for (int t = 0; t < TPW; ++t) {
            float part = 0.0f;
#pragma unroll
            for (int r = 0; r < 8; ++r) {
                part = fmaf(ftanh(d1[t][r]), sWout[b][16 * hh + r], part);
                part = fmaf(ftanh(d2[t][r]), sWout[b][16 * hh + 8 + r], part);
            }
            float u = part + __shfl_xor(part, 16, 32) + bO;   // SCALE=1, SHIFT=0
            acc[t]  = fmaf(u, w[t], acc[t]);
            wacc[t] += w[t];
        }
    };

    stage(0, 0);
    __syncthreads();

    // s-loop unrolled x2: buffer indices are literals in each body
    for (int s = 0; s < N_SUB; s += 2) {
        stage(s + 1, 1);                              // s+1 < N_SUB always (N_SUB even)
        if (s + 2 < N_SUB && tid < 32) {
            __builtin_prefetch(W_h1 + (s + 2) * HID * HID + tid * 32, 0, 1);
            __builtin_prefetch(W_h2 + (s + 2) * HID * HID + tid * 32, 0, 1);
        }
        compute(0);
        __syncthreads();

        if (s + 2 < N_SUB) stage(s + 2, 0);
        if (s + 3 < N_SUB && tid < 32) {
            __builtin_prefetch(W_h1 + (s + 3) * HID * HID + tid * 32, 0, 1);
            __builtin_prefetch(W_h2 + (s + 3) * HID * HID + tid * 32, 0, 1);
        }
        compute(1);
        __syncthreads();
    }

    // ---- normalization + hard BC; one writer per point (lower half) ----
    if (hh == 0) {
#pragma unroll
        for (int t = 0; t < TPW; ++t) {
            if (pv[t]) {
                float u   = acc[t] * frcp(wacc[t] + 1e-8f);
                float g   = -__sinf(3.14159265358979323846f * xxv[t]);
                float fac = ftanh(xxv[t] + 1.0f) * ftanh(xxv[t] - 1.0f) * ftanh(xt[t]);
                out[pp[t]] = fmaf(fac, u, g);
            }
        }
    }
}

extern "C" void kernel_launch(void* const* d_in, const int* in_sizes, int n_in,
                              void* d_out, int out_size, void* d_ws, size_t ws_size,
                              hipStream_t stream) {
    const float* x       = (const float*)d_in[0];
    const float* lo_core = (const float*)d_in[1];
    const float* hi_core = (const float*)d_in[2];
    const float* lo_ext  = (const float*)d_in[3];
    const float* hi_ext  = (const float*)d_in[4];
    const float* W_in    = (const float*)d_in[5];
    const float* b_in    = (const float*)d_in[6];
    const float* W_h1    = (const float*)d_in[7];
    const float* b_h1    = (const float*)d_in[8];
    const float* W_h2    = (const float*)d_in[9];
    const float* b_h2    = (const float*)d_in[10];
    const float* W_out   = (const float*)d_in[11];
    const float* b_out   = (const float*)d_in[12];
    float* out = (float*)d_out;

    const int n_pts = in_sizes[0] / 2;                   // N = 131072
    const int ptsPerBlock = 16 * TPW * (BLOCKT / 32);    // 512
    const int blocks = (n_pts + ptsPerBlock - 1) / ptsPerBlock;

    fbpinn_fused_wmma<<<blocks, BLOCKT, 0, stream>>>(
        x, lo_core, hi_core, lo_ext, hi_ext,
        W_in, b_in, W_h1, b_h1, W_h2, b_h2, W_out, b_out,
        out, n_pts);
}
